// DiffGraphTransformer_1726576857262
// MI455X (gfx1250) — matrix operand order repaired and checked by reading, not compiled, not run on hardware
//
#include <hip/hip_runtime.h>
#include <hip/hip_bf16.h>

// ---------------------------------------------------------------------------
// DiffGraphTransformer forward for MI455X (gfx1250, wave32, WMMA bf16)
//   T=1024, B=8, E=512, H=8, hd=64, scaling = 1/8
//   k = q (diff trick)  => only Q and V projections are needed.
//   Flash-style attention with multiplicative pe kernel folded into exp step.
//   Software-pipelined: next-stage fragment loads are issued before the
//   current stage's WMMAs with __builtin_amdgcn_sched_barrier(0) keeping the
//   scheduler from sinking them, so loads overlap matrix ops in-wave.
// ---------------------------------------------------------------------------

typedef __bf16 bf16_t;
typedef __attribute__((ext_vector_type(16))) __bf16 v16bf;
typedef __attribute__((ext_vector_type(8)))  __bf16 v8bf;
typedef __attribute__((ext_vector_type(8)))  float  v8f;

#define TSEQ  1024
#define BATCH 8
#define EMB   512
#define HEADS 8
#define HD    64
#define ROWS  (TSEQ * BATCH)      // 8192 tokens

// ---------------------------------------------------------------------------
// Fragment loaders (CDNA5 ISA 7.12.2 layouts)
// ---------------------------------------------------------------------------

// A-matrix 16x32 bf16: lane L holds row m = L%16;
//   elements j: k = (j/8)*16 + (L/16)*8 + j%8  -> two contiguous 8-elem groups.
__device__ __forceinline__ v16bf load_a_16x32(const bf16_t* __restrict__ base,
                                              int row0, int k0, int ld) {
  const int lane = threadIdx.x & 31;
  const int m = lane & 15, h = lane >> 4;
  const bf16_t* p = base + (size_t)(row0 + m) * ld + k0 + h * 8;
  v8bf lo = *(const v8bf*)(p);
  v8bf hi = *(const v8bf*)(p + 16);
  v16bf r;
#pragma unroll
  for (int i = 0; i < 8; ++i) { r[i] = lo[i]; r[i + 8] = hi[i]; }
  return r;
}

// B-matrix 32x16 bf16: lane L holds column n = L%16;
//   elements j: k = (j/8)*16 + (L/16)*8 + j%8  -> two contiguous 8-elem groups of B^T row n.
// 'base' is B transposed (row-major over n, k contiguous, leading dim ld).
__device__ __forceinline__ v16bf load_bt_32x16(const bf16_t* __restrict__ base,
                                               int n0, int k0, int ld) {
  const int lane = threadIdx.x & 31;
  const int n = lane & 15, h = lane >> 4;
  const bf16_t* p = base + (size_t)(n0 + n) * ld + k0 + h * 8;
  v8bf lo = *(const v8bf*)(p);
  v8bf hi = *(const v8bf*)(p + 16);
  v16bf r;
#pragma unroll
  for (int i = 0; i < 8; ++i) { r[i] = lo[i]; r[i + 8] = hi[i]; }
  return r;
}

__device__ __forceinline__ v8f wmma_bf16(v16bf a, v16bf b, v8f c) {
  return __builtin_amdgcn_wmma_f32_16x16x32_bf16(false, a, false, b,
                                                 (short)0, c, false, false);
}

__device__ __forceinline__ v8f vzero8() {
  v8f z = {0.f, 0.f, 0.f, 0.f, 0.f, 0.f, 0.f, 0.f};
  return z;
}

// ---------------------------------------------------------------------------
// Kernel 1: fp32 -> bf16 conversion of query, Wq, Wv, Wo
// ---------------------------------------------------------------------------
#define NX ((size_t)ROWS * EMB)       // 4194304
#define NW ((size_t)EMB * EMB)        // 262144

__global__ void convert_kernel(const float* __restrict__ query,
                               const float* __restrict__ Wi,   // (3E, E)
                               const float* __restrict__ Wout, // (E, E)
                               bf16_t* __restrict__ Xbf,
                               bf16_t* __restrict__ Wq,
                               bf16_t* __restrict__ Wv,
                               bf16_t* __restrict__ Wo) {
  size_t i = (size_t)blockIdx.x * blockDim.x + threadIdx.x;
  if (i < NX) { Xbf[i] = (bf16_t)query[i]; return; }
  size_t j = i - NX;
  if (j < NW) { Wq[j] = (bf16_t)Wi[j]; return; }            // rows [0, 512)
  j -= NW;
  if (j < NW) { Wv[j] = (bf16_t)Wi[2 * NW + j]; return; }   // rows [1024, 1536)
  j -= NW;
  if (j < NW) { Wo[j] = (bf16_t)Wout[j]; return; }
}

// ---------------------------------------------------------------------------
// Kernel 2: fused Q/V projection GEMM. One wave computes a 16x64 output strip.
//   Q -> Qh[bh][t][d] (row-major over t, d contiguous; serves as Q *and* K)
//   V -> Vt[bh][d][t] (transposed so B-fragments of w@V are contiguous)
// ---------------------------------------------------------------------------
__global__ __launch_bounds__(32) void proj_qv_kernel(
    const bf16_t* __restrict__ Xbf,   // (8192, 512)
    const bf16_t* __restrict__ Wq,    // (512, 512) row-major over k
    const bf16_t* __restrict__ Wv,
    const float*  __restrict__ bias,  // in_proj_bias (1536)
    bf16_t* __restrict__ Qh,          // (64, 1024, 64)
    bf16_t* __restrict__ Vt) {        // (64, 64, 1024)
  const int id       = blockIdx.x;
  const int rowTile  = id & 511;            // 8192/16 row tiles
  const int colStrip = (id >> 9) & 7;       // 512/64 column strips
  const int proj     = id >> 12;            // 0 = Q, 1 = V
  const bf16_t* W = proj ? Wv : Wq;
  const int r0 = rowTile * 16;
  const int e0 = colStrip * 64;

  v8f acc0 = vzero8(), acc1 = vzero8(), acc2 = vzero8(), acc3 = vzero8();

  // software pipeline: stage k0 resident, stage k0+32 in flight
  v16bf a  = load_a_16x32(Xbf, r0, 0, EMB);
  v16bf b0 = load_bt_32x16(W, e0 + 0,  0, EMB);
  v16bf b1 = load_bt_32x16(W, e0 + 16, 0, EMB);
  v16bf b2 = load_bt_32x16(W, e0 + 32, 0, EMB);
  v16bf b3 = load_bt_32x16(W, e0 + 48, 0, EMB);
#pragma unroll
  for (int k0 = 0; k0 < EMB; k0 += 32) {
    const int kn = (k0 + 32 < EMB) ? k0 + 32 : 0;   // last iter: dummy reload
    v16bf an  = load_a_16x32(Xbf, r0, kn, EMB);
    v16bf nb0 = load_bt_32x16(W, e0 + 0,  kn, EMB);
    v16bf nb1 = load_bt_32x16(W, e0 + 16, kn, EMB);
    v16bf nb2 = load_bt_32x16(W, e0 + 32, kn, EMB);
    v16bf nb3 = load_bt_32x16(W, e0 + 48, kn, EMB);
    __builtin_amdgcn_sched_barrier(0);   // keep next-stage loads above WMMAs
    acc0 = wmma_bf16(a, b0, acc0);
    acc1 = wmma_bf16(a, b1, acc1);
    acc2 = wmma_bf16(a, b2, acc2);
    acc3 = wmma_bf16(a, b3, acc3);
    a = an; b0 = nb0; b1 = nb1; b2 = nb2; b3 = nb3;
  }

  const int lane = threadIdx.x & 31;
  const int n = lane & 15, h = lane >> 4;
  const int biasBase = proj ? 2 * EMB : 0;
  v8f acc[4] = {acc0, acc1, acc2, acc3};
#pragma unroll
  for (int c = 0; c < 4; ++c) {
    const int e = e0 + c * 16 + n;
    const float bia = bias[biasBase + e];
    const int head = e >> 6, d = e & 63;
#pragma unroll
    for (int r = 0; r < 8; ++r) {
      const int R = r0 + r + h * 8;       // token index t*B + b
      const int t = R >> 3, bb = R & 7;
      const bf16_t o = (bf16_t)(acc[c][r] + bia);
      const size_t bh = (size_t)(bb * HEADS + head);
      if (proj == 0)
        Qh[(bh * TSEQ + t) * HD + d] = o;
      else
        Vt[(bh * HD + d) * TSEQ + t] = o;
    }
  }
}

// ---------------------------------------------------------------------------
// Kernel 3: flash-style diffusion attention. One wave owns (head bh, 16 query
// rows), keeps a 16x64 f32 accumulator + running max/sum, streams K/V in
// 32-column chunks, double-buffered in registers (stages A/B).
// pe is folded multiplicatively into the exp step (legal: it doesn't affect
// the max shift; the running sum rescales consistently).
// ---------------------------------------------------------------------------
__global__ __launch_bounds__(32) void attn_kernel(
    const bf16_t* __restrict__ Qh,   // (64, 1024, 64) — also the K matrix
    const bf16_t* __restrict__ Vt,   // (64, 64, 1024)
    const float*  __restrict__ pe,   // (8, 1024, 1024)
    bf16_t* __restrict__ Att) {      // (64, 1024, 64)
  __shared__ __align__(16) bf16_t pbuf[16 * 32];

  const int bh   = blockIdx.x >> 6;        // 64 head-batches
  const int tile = blockIdx.x & 63;        // 64 row tiles of 16
  const int t0   = tile * 16;
  const int b    = bh >> 3;                // bh = b*H + h  =>  pe batch
  const bf16_t* q   = Qh + (size_t)bh * TSEQ * HD;
  const bf16_t* vt  = Vt + (size_t)bh * HD * TSEQ;
  const float*  peb = pe + (size_t)b * TSEQ * TSEQ;

  const int lane = threadIdx.x & 31;
  const int n = lane & 15, h = lane >> 4;

  const v16bf aq0 = load_a_16x32(q, t0, 0,  HD);
  const v16bf aq1 = load_a_16x32(q, t0, 32, HD);

  v8f o0 = vzero8(), o1 = vzero8(), o2 = vzero8(), o3 = vzero8();
  float mrow[8], lrow[8];
#pragma unroll
  for (int r = 0; r < 8; ++r) { mrow[r] = -3.0e38f; lrow[r] = 0.f; }

  const float scaling = 0.125f;            // hd^-0.5 = 64^-0.5

  // ---- stage loader: all global traffic for one 32-column chunk ----------
  auto load_stage = [&](int s, v16bf& k00, v16bf& k01, v16bf& k10, v16bf& k11,
                        v16bf& vv0, v16bf& vv1, v16bf& vv2, v16bf& vv3,
                        float (&p0)[8], float (&p1)[8]) {
    k00 = load_bt_32x16(q, s,      0,  HD);
    k01 = load_bt_32x16(q, s,      32, HD);
    k10 = load_bt_32x16(q, s + 16, 0,  HD);
    k11 = load_bt_32x16(q, s + 16, 32, HD);
#pragma unroll
    for (int r = 0; r < 8; ++r) {
      const float* per = peb + (size_t)(t0 + r + h * 8) * TSEQ + s + n;
      p0[r] = per[0];
      p1[r] = per[16];
      if (s + 64 < TSEQ)   // keep the dominant HBM stream primed in L2
        __builtin_prefetch(per + 64, 0, 0);
    }
    vv0 = load_bt_32x16(vt, 0,  s, TSEQ);
    vv1 = load_bt_32x16(vt, 16, s, TSEQ);
    vv2 = load_bt_32x16(vt, 32, s, TSEQ);
    vv3 = load_bt_32x16(vt, 48, s, TSEQ);
  };

  // ---- per-chunk compute: scores, online softmax, LDS bounce, O += wV ----
  auto process = [&](const v16bf& k00, const v16bf& k01, const v16bf& k10,
                     const v16bf& k11, const v16bf& vv0, const v16bf& vv1,
                     const v16bf& vv2, const v16bf& vv3,
                     const float (&p0v)[8], const float (&p1v)[8]) {
    v8f sc0 = vzero8(), sc1 = vzero8();
    sc0 = wmma_bf16(aq0, k00, sc0);
    sc0 = wmma_bf16(aq1, k01, sc0);
    sc1 = wmma_bf16(aq0, k10, sc1);
    sc1 = wmma_bf16(aq1, k11, sc1);

    float corr[8];
#pragma unroll
    for (int r = 0; r < 8; ++r) {
      const float x0 = sc0[r] * scaling;
      const float x1 = sc1[r] * scaling;
      float tm = fmaxf(x0, x1);
      tm = fmaxf(tm, __shfl_xor(tm, 1, 32));
      tm = fmaxf(tm, __shfl_xor(tm, 2, 32));
      tm = fmaxf(tm, __shfl_xor(tm, 4, 32));
      tm = fmaxf(tm, __shfl_xor(tm, 8, 32));
      const float nm = fmaxf(mrow[r], tm);
      corr[r] = __expf(mrow[r] - nm);
      mrow[r] = nm;

      const float p0 = __expf(x0 - nm) * p0v[r];
      const float p1 = __expf(x1 - nm) * p1v[r];

      float rs = p0 + p1;
      rs += __shfl_xor(rs, 1, 32);
      rs += __shfl_xor(rs, 2, 32);
      rs += __shfl_xor(rs, 4, 32);
      rs += __shfl_xor(rs, 8, 32);
      lrow[r] = lrow[r] * corr[r] + rs;

      pbuf[(r + h * 8) * 32 + n]      = (bf16_t)p0;   // C layout -> row major
      pbuf[(r + h * 8) * 32 + 16 + n] = (bf16_t)p1;
    }
#pragma unroll
    for (int r = 0; r < 8; ++r) {
      o0[r] *= corr[r]; o1[r] *= corr[r]; o2[r] *= corr[r]; o3[r] *= corr[r];
    }
    const v16bf pa = load_a_16x32(pbuf, 0, 0, 32);
    o0 = wmma_bf16(pa, vv0, o0);
    o1 = wmma_bf16(pa, vv1, o1);
    o2 = wmma_bf16(pa, vv2, o2);
    o3 = wmma_bf16(pa, vv3, o3);
  };

  // ---- double-buffered main loop (2 chunks per iteration) ----------------
  v16bf k00a, k01a, k10a, k11a, v0a, v1a, v2a, v3a;
  v16bf k00b, k01b, k10b, k11b, v0b, v1b, v2b, v3b;
  float p0a[8], p1a[8], p0b[8], p1b[8];

  load_stage(0, k00a, k01a, k10a, k11a, v0a, v1a, v2a, v3a, p0a, p1a);
  for (int s0 = 0; s0 < TSEQ; s0 += 64) {
    load_stage(s0 + 32, k00b, k01b, k10b, k11b, v0b, v1b, v2b, v3b, p0b, p1b);
    __builtin_amdgcn_sched_barrier(0);
    process(k00a, k01a, k10a, k11a, v0a, v1a, v2a, v3a, p0a, p1a);

    const int sn = (s0 + 64 < TSEQ) ? s0 + 64 : 0;   // last iter: dummy reload
    load_stage(sn, k00a, k01a, k10a, k11a, v0a, v1a, v2a, v3a, p0a, p1a);
    __builtin_amdgcn_sched_barrier(0);
    process(k00b, k01b, k10b, k11b, v0b, v1b, v2b, v3b, p0b, p1b);
  }

  // normalize by clip(sum, 1e-6) and emit bf16 for the out-projection
#pragma unroll
  for (int r = 0; r < 8; ++r) {
    const float inv = 1.0f / fmaxf(lrow[r], 1e-6f);
    const int trow = t0 + r + h * 8;
    bf16_t* dst = Att + ((size_t)bh * TSEQ + trow) * HD;
    dst[n]      = (bf16_t)(o0[r] * inv);
    dst[16 + n] = (bf16_t)(o1[r] * inv);
    dst[32 + n] = (bf16_t)(o2[r] * inv);
    dst[48 + n] = (bf16_t)(o3[r] * inv);
  }
}

// ---------------------------------------------------------------------------
// Kernel 4: out-projection. A rows = tokens (t*B+b) gathered from the
// per-head Att layout (8-element k-groups never cross a 64-wide head chunk).
// ---------------------------------------------------------------------------
__device__ __forceinline__ v16bf load_a_att(const bf16_t* __restrict__ att,
                                            int r0, int k0) {
  const int lane = threadIdx.x & 31;
  const int m = lane & 15, h = lane >> 4;
  const int R = r0 + m;
  const int t = R >> 3, b = R & 7;
  v16bf out;
#pragma unroll
  for (int g = 0; g < 2; ++g) {
    const int k = k0 + g * 16 + h * 8;
    const int head = k >> 6, d = k & 63;
    const v8bf* p = (const v8bf*)(att +
        (((size_t)(b * HEADS + head)) * TSEQ + t) * HD + d);
    v8bf v = *p;
#pragma unroll
    for (int i = 0; i < 8; ++i) out[g * 8 + i] = v[i];
  }
  return out;
}

__global__ __launch_bounds__(32) void outproj_kernel(
    const bf16_t* __restrict__ Att,  // (64, 1024, 64)
    const bf16_t* __restrict__ Wo,   // (512, 512) row-major over k
    const float*  __restrict__ bias, // (512)
    float* __restrict__ Out) {       // (8192, 512) fp32
  const int id = blockIdx.x;
  const int rowTile  = id & 511;
  const int colStrip = id >> 9;
  const int r0 = rowTile * 16;
  const int e0 = colStrip * 64;

  v8f acc0 = vzero8(), acc1 = vzero8(), acc2 = vzero8(), acc3 = vzero8();

  v16bf a  = load_a_att(Att, r0, 0);
  v16bf b0 = load_bt_32x16(Wo, e0 + 0,  0, EMB);
  v16bf b1 = load_bt_32x16(Wo, e0 + 16, 0, EMB);
  v16bf b2 = load_bt_32x16(Wo, e0 + 32, 0, EMB);
  v16bf b3 = load_bt_32x16(Wo, e0 + 48, 0, EMB);
#pragma unroll
  for (int k0 = 0; k0 < EMB; k0 += 32) {
    const int kn = (k0 + 32 < EMB) ? k0 + 32 : 0;   // last iter: dummy reload
    v16bf an  = load_a_att(Att, r0, kn);
    v16bf nb0 = load_bt_32x16(Wo, e0 + 0,  kn, EMB);
    v16bf nb1 = load_bt_32x16(Wo, e0 + 16, kn, EMB);
    v16bf nb2 = load_bt_32x16(Wo, e0 + 32, kn, EMB);
    v16bf nb3 = load_bt_32x16(Wo, e0 + 48, kn, EMB);
    __builtin_amdgcn_sched_barrier(0);
    acc0 = wmma_bf16(a, b0, acc0);
    acc1 = wmma_bf16(a, b1, acc1);
    acc2 = wmma_bf16(a, b2, acc2);
    acc3 = wmma_bf16(a, b3, acc3);
    a = an; b0 = nb0; b1 = nb1; b2 = nb2; b3 = nb3;
  }

  const int lane = threadIdx.x & 31;
  const int n = lane & 15, h = lane >> 4;
  v8f acc[4] = {acc0, acc1, acc2, acc3};
#pragma unroll
  for (int c = 0; c < 4; ++c) {
    const int e = e0 + c * 16 + n;
    const float bia = bias[e];
#pragma unroll
    for (int r = 0; r < 8; ++r) {
      const int R = r0 + r + h * 8;
      Out[(size_t)R * EMB + e] = acc[c][r] + bia;
    }
  }
}

// ---------------------------------------------------------------------------
// Host launcher
// ---------------------------------------------------------------------------
extern "C" void kernel_launch(void* const* d_in, const int* in_sizes, int n_in,
                              void* d_out, int out_size, void* d_ws, size_t ws_size,
                              hipStream_t stream) {
  const float* query = (const float*)d_in[0];  // (1024, 8, 512)
  const float* pe    = (const float*)d_in[1];  // (8, 1024, 1024)
  const float* Wi    = (const float*)d_in[2];  // (1536, 512)
  const float* bi    = (const float*)d_in[3];  // (1536)
  const float* Wof   = (const float*)d_in[4];  // (512, 512)
  const float* bo    = (const float*)d_in[5];  // (512)
  float* out = (float*)d_out;                  // (1024, 8, 512)

  char* ws = (char*)d_ws;
  size_t off = 0;
  bf16_t* Xbf = (bf16_t*)(ws + off); off += NX * sizeof(bf16_t);            // 8 MB
  bf16_t* Wq  = (bf16_t*)(ws + off); off += NW * sizeof(bf16_t);            // 0.5 MB
  bf16_t* Wv  = (bf16_t*)(ws + off); off += NW * sizeof(bf16_t);
  bf16_t* Wo  = (bf16_t*)(ws + off); off += NW * sizeof(bf16_t);
  bf16_t* Qh  = (bf16_t*)(ws + off); off += (size_t)64 * TSEQ * HD * 2;     // 8 MB
  bf16_t* Vt  = (bf16_t*)(ws + off); off += (size_t)64 * HD * TSEQ * 2;     // 8 MB
  bf16_t* Att = (bf16_t*)(ws + off); off += (size_t)64 * TSEQ * HD * 2;     // 8 MB
  (void)ws_size; (void)in_sizes; (void)n_in; (void)out_size;

  // 1) fp32 -> bf16 staging
  {
    const size_t total = NX + 3 * NW;
    dim3 grid((unsigned)((total + 255) / 256));
    convert_kernel<<<grid, 256, 0, stream>>>(query, Wi, Wof, Xbf, Wq, Wv, Wo);
  }
  // 2) Q and V projections (K == Q, skipped)
  proj_qv_kernel<<<512 * 8 * 2, 32, 0, stream>>>(Xbf, Wq, Wv, bi, Qh, Vt);
  // 3) flash diffusion attention
  attn_kernel<<<64 * 64, 32, 0, stream>>>(Qh, Vt, pe, Att);
  // 4) out-projection straight into d_out
  outproj_kernel<<<512 * 8, 32, 0, stream>>>(Att, Wo, bo, out);
}
